// RoutingCapsules_11982958756249
// MI455X (gfx1250) — compile-verified, matrix operand order checked by reading, not run on hardware
//
#include <hip/hip_runtime.h>

// ---------------------------------------------------------------------------
// Capsule routing on MI455X (gfx1250, wave32, WMMA + TDM).
//
//   x: [B=32, Nin=2048, Din=16] f32   W: [Nin, Nout=64, Dout=32, Din=16] f32
//   u_hat[b,n,od] = sum_i W[n,od,i] * x[b,n,i]  ;  3 routing iterations.
//
// 3 fused passes, never materializing u_hat (512MB). Each pass streams W
// (256MB) through a TDM ring in LDS (tensor_load_to_lds + s_wait_tensorcnt,
// 3x32KB chunks in flight per WGP -> latency covered at 23.3TB/s), computes
// u_hat tiles with v_wmma_f32_16x16x32_f16 into LDS, and does routing with a
// shuffle-free diagonal agreement + per-lane register s accumulation.
// Deterministic per-chunk partials -> tiny reduce+squash kernels.
// ---------------------------------------------------------------------------

typedef __attribute__((ext_vector_type(16))) _Float16 v16h;
typedef __attribute__((ext_vector_type(8)))  float    v8f;
typedef __attribute__((ext_vector_type(4)))  unsigned int v4u;
typedef __attribute__((ext_vector_type(8)))  int      v8i;
typedef __attribute__((ext_vector_type(4)))  int      v4i;

#define B_TOT    32
#define NIN      2048
#define DIN      16
#define NOUT     64
#define DOUT     32
#define OD       2048                 // Nout*Dout
#define BBLK     16                   // b per workgroup (one b per wave)
#define NCHUNK   128                  // n-chunks (16 n each)
#define CHUNK    16
#define USTRIDE  2052                 // OD + 4 pad (stride%64==4 -> bank spread)

// --- TDM W-streaming ring ---
#define CROWS    512                  // W rows (od) per TDM chunk = 32KB global
#define CPN      (OD / CROWS)         // 4 chunks per n
#define TOTALC   (CHUNK * CPN)        // 64 chunks per workgroup per pass
#define RSTRIDE  20                   // DW per row in LDS: 16 data + 4 TDM pad
#define SLOTF    (CROWS * RSTRIDE)    // 10240 floats per ring slot (40KB)
#define NBUF     4
#define LA       3                    // chunks in flight (<= NBUF-1)

#define US_FLOATS (BBLK * USTRIDE)                   // 32832
#define XS_OFF    (US_FLOATS * 4)                    // 131328 B
#define RING_OFF  (XS_OFF + BBLK * DIN * 2)          // 131840 B (16B aligned)
#define SMEM_BYTES (RING_OFF + NBUF * SLOTF * 4)     // 295680 B  (< 320KB WGP)

__device__ __forceinline__ float wave_sum(float v) {
#pragma unroll
  for (int off = 16; off >= 1; off >>= 1) v += __shfl_xor(v, off, 32);
  return v;
}
__device__ __forceinline__ float wave_max(float v) {
#pragma unroll
  for (int off = 16; off >= 1; off >>= 1) v = fmaxf(v, __shfl_xor(v, off, 32));
  return v;
}

// Issue one TDM DMA: 8192 contiguous f32 (512 W rows) -> LDS ring slot,
// padding 4 DW after every 16 DW (row stride 20 DW: bank-spread + 16B align).
// D# per CDNA5 ISA 8.3/8.4; tracked by TENSORcnt.
__device__ __forceinline__ void tdm_load_chunk(const float* gsrc, unsigned lds_off) {
  const unsigned long long ga = (unsigned long long)(uintptr_t)gsrc;
  const unsigned ne = CROWS * DIN;                       // 8192 elements
  v4u g0 = { 1u,                                         // count=1 (user D#)
             lds_off,                                    // lds_addr (bytes)
             (unsigned)ga,                               // global_addr[31:0]
             (unsigned)(ga >> 32) | 0x80000000u };       // ga[56:32] | type=2
  v8i g1 = { (int)((2u << 16)      // data_size = 4B
                 | (1u << 20)      // pad_enable
                 | (3u << 22)      // pad_interval code 3 = every 16 DW
                 | (3u << 25)),    // pad_amount  code 3 = 4 DW
             (int)((ne & 0xffffu) << 16),   // tensor_dim0 lo16  (bits 63:48)
             (int)((ne >> 16) | (1u << 16)),// tensor_dim0 hi16 | tensor_dim1=1
             (int)((ne & 0xffffu) << 16),   // tensor_dim1 hi=0 | tile_dim0=8192
             1,                              // tile_dim1=1, tile_dim2=0
             (int)ne,                        // tensor_dim0_stride lo32
             0, 0 };
  // 6-arg form (clang-23 / therock headers): groups 0..3 + extra group + cpol
  __builtin_amdgcn_tensor_load_to_lds(g0, g1, (v4i)0, (v4i)0, (v8i)0, 0);
}

// One fused pass. IT=0: c uniform. IT=1: b1=agree(v0), store blog.
// IT=2: b2=blog+agree(v1). Workgroup: 512 threads = 16 waves; wave w owns
// b-local w; (bblk,chunk) grid = (2,128); one workgroup per WGP.
template <int IT>
__global__ __launch_bounds__(512) void caps_pass_kernel(
    const float* __restrict__ x, const float* __restrict__ W,
    const float* __restrict__ vprev,   // [B,NOUT,DOUT] (null for IT=0)
    float* __restrict__ blog,          // [B,NIN,NOUT]
    float* __restrict__ part)          // [2*NCHUNK][BBLK][OD]
{
  const int bblk  = blockIdx.x;        // 0..1 (adjacent twins share W via L2)
  const int chunk = blockIdx.y;        // 0..127
  const int tid   = threadIdx.x;
  const int wv    = tid >> 5;          // 0..15
  const int lane  = tid & 31;
  const int bcol  = lane & 15;         // WMMA row/col within half-wave
  const int kh    = lane >> 4;         // K-half (0: K0..7, 1: K8..15)

  extern __shared__ char smemraw[];
  float*    uS   = (float*)smemraw;                   // [BBLK][USTRIDE]
  _Float16* xs   = (_Float16*)(smemraw + XS_OFF);     // [BBLK][DIN]
  float*    ring = (float*)(smemraw + RING_OFF);      // [NBUF][SLOTF]

  // W chunk g (0..TOTALC-1) for this workgroup: n-local = g/CPN, row-block g%CPN
  const float* wbase = W + (size_t)chunk * CHUNK * OD * DIN;

  // ---- prime the TDM pipeline ----
  if (wv == 0) {
#pragma unroll
    for (int p = 0; p < LA; ++p)
      tdm_load_chunk(wbase + (size_t)p * CROWS * DIN,
                     (unsigned)RING_OFF + (unsigned)(p % NBUF) * (SLOTF * 4));
  }

  float sreg[NOUT];                    // s partial: lane=d, index=o
#pragma unroll
  for (int o = 0; o < NOUT; ++o) sreg[o] = 0.0f;

  int g = 0;
  for (int nn = 0; nn < CHUNK; ++nn) {
    const int n = chunk * CHUNK + nn;

    // stage x[n] for our 16 b (f32 -> f16); visible after next barrier
    if (tid < BBLK * DIN) {
      const int bl = tid >> 4, i = tid & 15;
      xs[bl * DIN + i] =
          (_Float16)x[((size_t)(bblk * BBLK + bl) * NIN + n) * DIN + i];
    }

    for (int c = 0; c < CPN; ++c, ++g) {
      // chunk g ready? (wave0 owns TENSORcnt); barrier publishes to all waves
      if (wv == 0) {
        if (g + LA <= TOTALC) __builtin_amdgcn_s_wait_tensorcnt(LA - 1);
        else                  __builtin_amdgcn_s_wait_tensorcnt(0);
      }
      __syncthreads();
      // keep LA chunks in flight (slot safe: prior occupant g+LA-NBUF <= g-1
      // was consumed before a barrier wave0 has already crossed)
      if (wv == 0 && g + LA < TOTALC)
        tdm_load_chunk(wbase + (size_t)(g + LA) * CROWS * DIN,
                       (unsigned)RING_OFF + (unsigned)((g + LA) % NBUF) * (SLOTF * 4));

      // B fragment (x side), rebuilt cheaply per chunk from LDS
      v16h bfrag = (v16h)0;
      {
        const _Float16* xp = xs + bcol * DIN + kh * 8;
#pragma unroll
        for (int j = 0; j < 8; ++j) bfrag[j] = xp[j];   // K=16..31 stay 0
      }

      const float* slot = ring + (size_t)(g % NBUF) * SLOTF;
#pragma unroll
      for (int tt = 0; tt < 2; ++tt) {                  // 32 tiles / 16 waves
        const int tl = wv * 2 + tt;                     // tile in chunk
        const int rowl = tl * 16 + bcol;                // row in chunk
        const float* wr = slot + rowl * RSTRIDE + kh * 8;
        const float4 w0 = *(const float4*)(wr);
        const float4 w1 = *(const float4*)(wr + 4);
        v16h afrag = (v16h)0;
        afrag[0] = (_Float16)w0.x; afrag[1] = (_Float16)w0.y;
        afrag[2] = (_Float16)w0.z; afrag[3] = (_Float16)w0.w;
        afrag[4] = (_Float16)w1.x; afrag[5] = (_Float16)w1.y;
        afrag[6] = (_Float16)w1.z; afrag[7] = (_Float16)w1.w;

        v8f acc = (v8f)0;
        acc = __builtin_amdgcn_wmma_f32_16x16x32_f16(
            false, afrag, false, bfrag, (short)0, acc, false, false);

        const int od0 = c * CROWS + tl * 16 + kh * 8;   // C rows this lane holds
        float* up = uS + bcol * USTRIDE + od0;          // 16B aligned
        *(float4*)(up)     = make_float4(acc[0], acc[1], acc[2], acc[3]);
        *(float4*)(up + 4) = make_float4(acc[4], acc[5], acc[6], acc[7]);
      }
    }
    __syncthreads();   // full u[16b][2048od] tile visible

    // ---- routing: wave owns b-local wv; lane = d; lane L owns o=L, o=L+32 ----
    const int b = bblk * BBLK + wv;
    const float* ub = uS + wv * USTRIDE;
    float c0 = 1.0f / 64.0f, c1 = 1.0f / 64.0f;

    if (IT > 0) {
      float lo0 = 0.0f, lo1 = 0.0f;
      if (IT == 2) {
        const float* bp = blog + ((size_t)b * NIN + n) * NOUT;
        lo0 = bp[lane]; lo1 = bp[lane + 32];
      }
      const float* vp = vprev + (size_t)b * OD;
      // diagonal walk: zero shuffles, conflict-free LDS banks
#pragma unroll
      for (int k = 0; k < 32; ++k) {
        const int d = (lane + k) & 31;
        lo0 += ub[lane * 32 + d]        * vp[lane * 32 + d];
        lo1 += ub[(lane + 32) * 32 + d] * vp[(lane + 32) * 32 + d];
      }
      if (IT == 1) {
        float* bp = blog + ((size_t)b * NIN + n) * NOUT;
        bp[lane] = lo0; bp[lane + 32] = lo1;
      }
      const float m  = wave_max(fmaxf(lo0, lo1));
      const float e0 = __expf(lo0 - m), e1 = __expf(lo1 - m);
      const float inv = 1.0f / wave_sum(e0 + e1);
      c0 = e0 * inv; c1 = e1 * inv;
    }
#pragma unroll
    for (int o = 0; o < NOUT; ++o) {                    // s += c_o * u_o
      const float co = (IT == 0) ? (1.0f / 64.0f)
                                 : __shfl((o < 32) ? c0 : c1, o & 31, 32);
      sreg[o] += co * ub[o * 32 + lane];
    }
  }

  // deterministic partials: [ (bblk*NCHUNK+chunk) ][ b-local ][ od ]
  float* pp = part + ((size_t)(bblk * NCHUNK + chunk) * BBLK + wv) * OD;
#pragma unroll
  for (int o = 0; o < NOUT; ++o) pp[o * 32 + lane] = sreg[o];
}

// Reduce partials over chunks, then squash: v = (sq/(1+sq)) * s / sqrt(sq+eps)
__global__ __launch_bounds__(32) void caps_reduce_squash(
    const float* __restrict__ part, float* __restrict__ vout)
{
  const int o = blockIdx.x, b = blockIdx.y, d = threadIdx.x;
  const int bblk = b >> 4, bl = b & 15;
  float s = 0.0f;
  for (int c = 0; c < NCHUNK; ++c)
    s += part[((size_t)(bblk * NCHUNK + c) * BBLK + bl) * OD + o * 32 + d];
  const float sq = wave_sum(s * s);
  const float scale = (sq / (1.0f + sq)) * rsqrtf(sq + 1e-8f);
  vout[((size_t)b * NOUT + o) * DOUT + d] = scale * s;
}

extern "C" void kernel_launch(void* const* d_in, const int* in_sizes, int n_in,
                              void* d_out, int out_size, void* d_ws, size_t ws_size,
                              hipStream_t stream) {
  const float* x = (const float*)d_in[0];   // [32,2048,16]
  const float* W = (const float*)d_in[1];   // [1,2048,64,32,16]
  float* out = (float*)d_out;               // [32,64,32]

  float* ws   = (float*)d_ws;
  float* part = ws;                                    // 2*128*16*2048 f32 (32MB)
  float* v0   = part + (size_t)2 * NCHUNK * BBLK * OD;
  float* v1   = v0 + (size_t)B_TOT * OD;
  float* blog = v1 + (size_t)B_TOT * OD;               // 32*2048*64 f32 (16MB)

  const dim3 pgrid(2, NCHUNK);    // bblk fastest -> W-twin L2 reuse
  const dim3 rgrid(NOUT, B_TOT);

  caps_pass_kernel<0><<<pgrid, 512, SMEM_BYTES, stream>>>(x, W, nullptr, blog, part);
  caps_reduce_squash<<<rgrid, 32, 0, stream>>>(part, v0);
  caps_pass_kernel<1><<<pgrid, 512, SMEM_BYTES, stream>>>(x, W, v0, blog, part);
  caps_reduce_squash<<<rgrid, 32, 0, stream>>>(part, v1);
  caps_pass_kernel<2><<<pgrid, 512, SMEM_BYTES, stream>>>(x, W, v1, blog, part);
  caps_reduce_squash<<<rgrid, 32, 0, stream>>>(part, out);
}
